// M18RelationalInterpreter_32899449487391
// MI455X (gfx1250) — matrix operand order repaired and checked by reading, not compiled, not run on hardware
//
#include <hip/hip_runtime.h>
#include <hip/hip_bf16.h>
#include <math.h>

#define Bn 4
#define Kn 128
#define Hn 896
#define Rn 8
#define BK (Bn * Kn)          // 512 rows of E
#define KSTEPS (Hn / 32)      // 28 WMMA k-steps
#define MT 4                  // 4 m-tiles (64 pairs) per workgroup

typedef __bf16 bf16_t;
typedef __attribute__((ext_vector_type(16))) __bf16 v16bf;
typedef __attribute__((ext_vector_type(8)))  __bf16 v8bf;
typedef __attribute__((ext_vector_type(8)))  float  v8f;

// ---------------------------------------------------------------------------
// Kernel 1: precision conversion + weight transposes.
//   Ebf   [BK][H]  = bf16(E)
//   W1cT  [n][k]   = bf16(W1c[k][n])          (K-contiguous for B-fragments)
//   WsumT [n][k]   = bf16(W1a[k][n] + W1d[k][n])
//   WdiffT[n][k]   = bf16(W1b[k][n] - W1d[k][n])
// ---------------------------------------------------------------------------
__global__ __launch_bounds__(256) void prep_kernel(
    const float* __restrict__ E, const float* __restrict__ W1,
    bf16_t* __restrict__ Ebf, bf16_t* __restrict__ W1cT,
    bf16_t* __restrict__ WsumT, bf16_t* __restrict__ WdiffT) {
  const int idx = blockIdx.x * blockDim.x + threadIdx.x;
  if (idx < BK * Hn) Ebf[idx] = (bf16_t)E[idx];
  if (idx < Hn * Hn) {
    const int n = idx / Hn, k = idx % Hn;
    const float a = W1[(size_t)k * Hn + n];
    const float bb = W1[(size_t)(Hn + k) * Hn + n];
    const float c = W1[(size_t)(2 * Hn + k) * Hn + n];
    const float d = W1[(size_t)(3 * Hn + k) * Hn + n];
    W1cT[idx] = (bf16_t)c;
    WsumT[idx] = (bf16_t)(a + d);
    WdiffT[idx] = (bf16_t)(bb - d);
  }
}

// 16-bit A layout: lane holds row (lane&15); element e -> K = (e>=8?16:0)+(lane>=16?8:0)+(e&7)
__device__ __forceinline__ v16bf load_a_rows(const bf16_t* __restrict__ row,
                                             int kb, int half) {
  const v8bf a0 = *(const v8bf*)(row + kb + half * 8);
  const v8bf a1 = *(const v8bf*)(row + kb + 16 + half * 8);
  v16bf a;
#pragma unroll
  for (int e = 0; e < 8; ++e) { a[e] = a0[e]; a[e + 8] = a1[e]; }
  return a;
}

// 16-bit B layout from K-contiguous weights: lane holds col (lane&15); element e -> K = (e>=8?16:0)+(lane>=16?8:0)+(e&7)
__device__ __forceinline__ v16bf load_b_frag(const bf16_t* __restrict__ WT,
                                             int n, int kb, int half) {
  const bf16_t* p = WT + (size_t)n * Hn + kb + half * 8;
  const v8bf b0 = *(const v8bf*)(p);
  const v8bf b1 = *(const v8bf*)(p + 16);
  v16bf bf;
#pragma unroll
  for (int e = 0; e < 8; ++e) { bf[e] = b0[e]; bf[e + 8] = b1[e]; }
  return bf;
}

// ---------------------------------------------------------------------------
// Kernel 2: term_i = E @ (W1a+W1d), term_j = E @ (W1b-W1d) via WMMA bf16.
// ---------------------------------------------------------------------------
__global__ __launch_bounds__(256) void term_kernel(
    const bf16_t* __restrict__ Ebf, const bf16_t* __restrict__ WsumT,
    const bf16_t* __restrict__ WdiffT, float* __restrict__ term_i,
    float* __restrict__ term_j) {
  const int wg = blockIdx.x;
  const int which = wg & 1;
  const int m0 = (wg >> 1) * 16;
  const bf16_t* WT = which ? WdiffT : WsumT;
  float* outT = which ? term_j : term_i;

  const int lane = threadIdx.x & 31;
  const int wave = threadIdx.x >> 5;
  const int half = lane >> 4;
  const int lrow = lane & 15;
  const int n0 = wave * 112;

  const bf16_t* Arow = Ebf + (size_t)(m0 + lrow) * Hn;

  v8f acc[7] = {};
  for (int ks = 0; ks < KSTEPS; ++ks) {
    const int kb = ks * 32;
    const v16bf a = load_a_rows(Arow, kb, half);
#pragma unroll
    for (int t = 0; t < 7; ++t) {
      const v16bf bf = load_b_frag(WT, n0 + t * 16 + lrow, kb, half);
      acc[t] = __builtin_amdgcn_wmma_f32_16x16x32_bf16(
          false, a, false, bf, (short)0, acc[t], false, false);
    }
  }
#pragma unroll
  for (int t = 0; t < 7; ++t) {
    const int n = n0 + t * 16 + lrow;
#pragma unroll
    for (int v = 0; v < 8; ++v) {
      const int m = m0 + v + half * 8;          // C/D layout: M = v + 8*(lane>=16)
      outT[(size_t)m * Hn + n] = acc[t][v];
    }
  }
}

// ---------------------------------------------------------------------------
// Kernel 3: fused main GEMM + bias + ReLU + (x W2 + b2) + softmax.
// One workgroup per (b, i, 64-j block): wave tile Mt=64 x Nt=112
// -> ~37 FLOP per L2 byte; W1cT traffic drops 4x vs a 16-pair tile.
// Epilogue streams the four 16-pair chunks through one 57.6KB LDS h-buffer.
// ---------------------------------------------------------------------------
#define HSTR 900  // 896 % 64banks == 0 -> pad to avoid half-wave bank aliasing

__global__ __launch_bounds__(256) void fused_kernel(
    const bf16_t* __restrict__ Ebf, const bf16_t* __restrict__ W1cT,
    const float* __restrict__ term_i, const float* __restrict__ term_j,
    const float* __restrict__ b1, const float* __restrict__ W2,
    const float* __restrict__ b2, float* __restrict__ out) {
  const int tb = blockIdx.x;            // 0..1023
  const int b = tb >> 8;                // 128 i * 2 jq per batch
  const int i = (tb >> 1) & (Kn - 1);
  const int j0 = (tb & 1) * 64;

  __shared__ float hbuf[16 * HSTR];     // 57.6 KB relu'd hidden chunk
  __shared__ float part[256];
  __shared__ float lsm[16 * Rn];

  const int lane = threadIdx.x & 31;
  const int wave = threadIdx.x >> 5;
  const int half = lane >> 4;
  const int lrow = lane & 15;
  const int n0 = wave * 112;

  const bf16_t* Ei = Ebf + ((size_t)b * Kn + i) * Hn;
  const bf16_t* Ej[MT];
#pragma unroll
  for (int mt = 0; mt < MT; ++mt)
    Ej[mt] = Ebf + ((size_t)b * Kn + j0 + mt * 16 + lrow) * Hn;

  v8f acc[MT][7] = {};
  for (int ks = 0; ks < KSTEPS; ++ks) {
    const int kb = ks * 32;
    // Shared Ei slices for this k-step
    const v8bf ei0 = *(const v8bf*)(Ei + kb + half * 8);
    const v8bf ei1 = *(const v8bf*)(Ei + kb + 16 + half * 8);
    // Build 4 A fragments: rows are Ei (.) Ej (v_pk_mul_bf16-able)
    v16bf a[MT];
#pragma unroll
    for (int mt = 0; mt < MT; ++mt) {
      const v8bf ej0 = *(const v8bf*)(Ej[mt] + kb + half * 8);
      const v8bf ej1 = *(const v8bf*)(Ej[mt] + kb + 16 + half * 8);
      const v8bf p0 = ei0 * ej0;
      const v8bf p1 = ei1 * ej1;
#pragma unroll
      for (int e = 0; e < 8; ++e) { a[mt][e] = p0[e]; a[mt][e + 8] = p1[e]; }
    }
    // Each B fragment feeds 4 WMMAs (4x reuse)
#pragma unroll
    for (int nt = 0; nt < 7; ++nt) {
      const v16bf bf = load_b_frag(W1cT, n0 + nt * 16 + lrow, kb, half);
#pragma unroll
      for (int mt = 0; mt < MT; ++mt) {
        acc[mt][nt] = __builtin_amdgcn_wmma_f32_16x16x32_bf16(
            false, a[mt], false, bf, (short)0, acc[mt][nt], false, false);
      }
    }
  }

  // Epilogue: stream 4 chunks of 16 pairs through LDS
  const float* ti = term_i + ((size_t)b * Kn + i) * Hn;
  for (int mt = 0; mt < MT; ++mt) {
#pragma unroll
    for (int nt = 0; nt < 7; ++nt) {
      const int n = n0 + nt * 16 + lrow;
      const float tin = ti[n];
      const float b1n = b1[n];
#pragma unroll
      for (int v = 0; v < 8; ++v) {
        const int m = v + half * 8;    // pair row within chunk
        const float tj = term_j[((size_t)b * Kn + j0 + mt * 16 + m) * Hn + n];
        float h = acc[mt][nt][v] + tin + tj + b1n;
        hbuf[m * HSTR + n] = h > 0.f ? h : 0.f;
      }
    }
    __syncthreads();

    // logits for this chunk: 16 pairs x 8 heads; split-K over 2 halves of H
    {
      const int r = threadIdx.x & 7;
      const int m = (threadIdx.x >> 3) & 15;
      const int seg = threadIdx.x >> 7;
      const float* hrow = &hbuf[m * HSTR + seg * (Hn / 2)];
      const float* w2p = W2 + (size_t)seg * (Hn / 2) * Rn + r;
      float s = 0.f;
      for (int n = 0; n < Hn / 2; ++n) s = fmaf(hrow[n], w2p[n * Rn], s);
      part[threadIdx.x] = s;
    }
    __syncthreads();
    if (threadIdx.x < 128) {
      const int r = threadIdx.x & 7;
      lsm[threadIdx.x] = part[threadIdx.x] + part[threadIdx.x + 128] + b2[r];
    }
    __syncthreads();
    if (threadIdx.x < 16) {
      const int m = threadIdx.x;
      float v[Rn], mx = -3.4e38f;
#pragma unroll
      for (int r = 0; r < Rn; ++r) { v[r] = lsm[m * Rn + r]; mx = fmaxf(mx, v[r]); }
      float ssum = 0.f;
#pragma unroll
      for (int r = 0; r < Rn; ++r) { v[r] = __expf(v[r] - mx); ssum += v[r]; }
      const float inv = 1.f / ssum;
      float* o = out + (((size_t)b * Kn + i) * Kn + (j0 + mt * 16 + m)) * Rn;
#pragma unroll
      for (int r = 0; r < Rn; ++r) o[r] = v[r] * inv;
    }
    __syncthreads();
  }
}

// ---------------------------------------------------------------------------
extern "C" void kernel_launch(void* const* d_in, const int* in_sizes, int n_in,
                              void* d_out, int out_size, void* d_ws,
                              size_t ws_size, hipStream_t stream) {
  (void)in_sizes; (void)n_in; (void)out_size; (void)ws_size;
  const float* E  = (const float*)d_in[0];
  const float* W1 = (const float*)d_in[1];
  const float* b1 = (const float*)d_in[2];
  const float* W2 = (const float*)d_in[3];
  const float* b2 = (const float*)d_in[4];
  float* out = (float*)d_out;

  char* ws = (char*)d_ws;
  size_t off = 0;
  bf16_t* Ebf    = (bf16_t*)(ws + off); off += (size_t)BK * Hn * sizeof(bf16_t);
  bf16_t* W1cT   = (bf16_t*)(ws + off); off += (size_t)Hn * Hn * sizeof(bf16_t);
  bf16_t* WsumT  = (bf16_t*)(ws + off); off += (size_t)Hn * Hn * sizeof(bf16_t);
  bf16_t* WdiffT = (bf16_t*)(ws + off); off += (size_t)Hn * Hn * sizeof(bf16_t);
  float*  term_i = (float*)(ws + off);  off += (size_t)BK * Hn * sizeof(float);
  float*  term_j = (float*)(ws + off);  off += (size_t)BK * Hn * sizeof(float);

  prep_kernel<<<(Hn * Hn + 255) / 256, 256, 0, stream>>>(E, W1, Ebf, W1cT,
                                                         WsumT, WdiffT);
  term_kernel<<<64, 256, 0, stream>>>(Ebf, WsumT, WdiffT, term_i, term_j);
  fused_kernel<<<Bn * Kn * (Kn / 64), 256, 0, stream>>>(
      Ebf, W1cT, term_i, term_j, b1, W2, b2, out);
}
